// RNNDecoder_46823733461343
// MI455X (gfx1250) — matrix operand order repaired and checked by reading, not compiled, not run on hardware
//
#include <hip/hip_runtime.h>

typedef __bf16 bf16;
typedef __attribute__((ext_vector_type(16))) __bf16 v16bf;
typedef __attribute__((ext_vector_type(8)))  float  v8f;

constexpr int cB = 64, cT = 64, cS = 128, cD = 512, cH = 512;
constexpr int cV = 32004, cVP = 32064;          // vocab, padded to /64
constexpr int cG = 3 * cH;                      // 1536 gate width
constexpr int cKX = cD + cH;                    // 1024 concat input

// ---------------------------------------------------------------- utilities
__global__ void k_cvt_bf16(const float* __restrict__ src, bf16* __restrict__ dst, int n) {
    int i = blockIdx.x * blockDim.x + threadIdx.x;
    if (i < n) dst[i] = (bf16)src[i];
}

// Convert W2 [rows,cols] f32 -> [rows_pad,cols] bf16, zero-filling pad rows.
__global__ void k_cvt_bf16_pad(const float* __restrict__ src, bf16* __restrict__ dst,
                               int rows, int rows_pad, int cols) {
    int i = blockIdx.x * blockDim.x + threadIdx.x;
    if (i < rows_pad * cols) {
        int r = i / cols;
        dst[i] = (r < rows) ? (bf16)src[i] : (bf16)0.0f;
    }
}

__global__ void k_init_state(const float* __restrict__ st,
                             float* __restrict__ h0, float* __restrict__ h1) {
    int i = blockIdx.x * blockDim.x + threadIdx.x;
    if (i < cB * cH) { h0[i] = st[i]; h1[i] = st[cB * cH + i]; }
}

// -------------------------------------------------- attention + concat(x_t, wm)
// One block per batch element; 128 threads = one per source position (S=128).
__global__ void __launch_bounds__(128) k_attention_concat(
        const float* __restrict__ inp, const float* __restrict__ mem,
        const float* __restrict__ h1, float* __restrict__ x0, int t) {
    int b = blockIdx.x, tid = threadIdx.x;
    __shared__ float sh[cH];
    __shared__ float sw[cS];
    __shared__ float red[cS];

    for (int i = tid; i < cH; i += 128) sh[i] = h1[b * cH + i];
    __syncthreads();

    // scores[s] = memory[b,s,:] . h1[b,:]
    const float* mrow = mem + ((size_t)b * cS + tid) * cH;
    float acc = 0.f;
    for (int h = 0; h < cH; ++h) acc += mrow[h] * sh[h];
    sw[tid] = acc; red[tid] = acc;
    __syncthreads();
    for (int s = 64; s > 0; s >>= 1) {
        if (tid < s) red[tid] = fmaxf(red[tid], red[tid + s]);
        __syncthreads();
    }
    float mx = red[0];
    __syncthreads();
    float e = expf(sw[tid] - mx);
    sw[tid] = e; red[tid] = e;
    __syncthreads();
    for (int s = 64; s > 0; s >>= 1) {
        if (tid < s) red[tid] += red[tid + s];
        __syncthreads();
    }
    float inv = 1.f / red[0];
    __syncthreads();
    sw[tid] *= inv;
    __syncthreads();

    // weighted memory -> x0[b, 512:1024]
    for (int h = tid; h < cH; h += 128) {
        float a = 0.f;
        for (int s = 0; s < cS; ++s) a += sw[s] * mem[((size_t)b * cS + s) * cH + h];
        x0[b * cKX + cD + h] = a;
    }
    // x_t -> x0[b, 0:512]
    for (int d = tid; d < cD; d += 128)
        x0[b * cKX + d] = inp[((size_t)b * cT + t) * cD + d];
}

// ------------------------------------------------------------- WMMA GEMM tile
// C[M,N] = A[M,K](f32) * W[N,K](bf16)^T (+ bias).
// Block = 128 threads = 4 waves; block tile 64(M) x 64(N).
// Wave w computes rows [blk*64 + w*16, +16) over all 64 columns (4 accumulators,
// A-fragment reused 4x). B tile (32K-deep slice, [n][k] = 64x32 bf16 = 4 KB) is
// staged in LDS once per block per k-slice via global_load_async_to_lds_b128
// (ASYNCcnt) and consumed by all 4 waves through DS loads.
__global__ void __launch_bounds__(128) k_wmma_gemm(
        const float* __restrict__ A, int lda,
        const bf16* __restrict__ W, int ldw,
        float* __restrict__ C, int ldc,
        int K, int Ntot, const float* __restrict__ bias) {
    __shared__ __align__(16) bf16 sB[64 * 32];   // [n][k] tile

    const int tid  = threadIdx.x;
    const int wv   = tid >> 5;        // wave id 0..3 -> m sub-tile
    const int lane = tid & 31;
    const int half = lane >> 4;       // 16-lane half of the wave
    const int l    = lane & 15;

    const int mblk = blockIdx.x * 64 + wv * 16;
    const int n0   = blockIdx.y * 64;

    v8f acc0 = {}, acc1 = {}, acc2 = {}, acc3 = {};

    const float*   Arow    = A + (size_t)(mblk + l) * lda;
    const uint32_t ldsBase = (uint32_t)(uintptr_t)(&sB[0]);

    for (int k0 = 0; k0 < K; k0 += 32) {
        __syncthreads();              // previous slice fully consumed
        // stage B tile rows n0..n0+63, cols k0..k0+31 into sB[n][k]
#pragma unroll
        for (int i = 0; i < 2; ++i) {
            int c    = tid + i * 128;         // 16-byte chunk id, 256 total
            int nrow = c >> 2;                // 4 chunks per 64B row
            int kk   = (c & 3) * 8;           // bf16 offset within row
            uint64_t ga = (uint64_t)(uintptr_t)(W + (size_t)(n0 + nrow) * ldw + k0 + kk);
            uint32_t la = ldsBase + (uint32_t)c * 16;
            asm volatile("global_load_async_to_lds_b128 %0, %1, off"
                         :: "v"(la), "v"(ga) : "memory");
        }
        asm volatile("s_wait_asynccnt 0" ::: "memory");
        __syncthreads();              // B tile visible to all waves

        // A fragment (f32 -> bf16 in-register); ISA 7.12.2 16-bit A 16x32:
        // e=0..7 -> K=half*8+e ; e=8..15 -> K=16+half*8+(e-8)
        v16bf a;
#pragma unroll
        for (int e = 0; e < 16; ++e) {
            int ka = k0 + half * 8 + e + ((e >= 8) ? 8 : 0);
            a[e] = (bf16)Arow[ka];
        }
        // B fragments from LDS: lane = column (nf*16+l), e=0..7 -> K=half*8+e ; e=8..15 -> K=16+half*8+(e-8)
        v16bf b0, b1, b2, b3;
#pragma unroll
        for (int e = 0; e < 16; ++e) {
            int ko = half * 8 + e + ((e >= 8) ? 8 : 0);
            b0[e] = sB[(0 * 16 + l) * 32 + ko];
            b1[e] = sB[(1 * 16 + l) * 32 + ko];
            b2[e] = sB[(2 * 16 + l) * 32 + ko];
            b3[e] = sB[(3 * 16 + l) * 32 + ko];
        }
        acc0 = __builtin_amdgcn_wmma_f32_16x16x32_bf16(false, a, false, b0, (short)0, acc0, false, false);
        acc1 = __builtin_amdgcn_wmma_f32_16x16x32_bf16(false, a, false, b1, (short)0, acc1, false, false);
        acc2 = __builtin_amdgcn_wmma_f32_16x16x32_bf16(false, a, false, b2, (short)0, acc2, false, false);
        acc3 = __builtin_amdgcn_wmma_f32_16x16x32_bf16(false, a, false, b3, (short)0, acc3, false, false);
    }

    // C/D layout: lane = N (within 16), VGPR v = M within half (ISA 7.12.2)
    v8f accs[4] = {acc0, acc1, acc2, acc3};
#pragma unroll
    for (int nf = 0; nf < 4; ++nf) {
        int nn = n0 + nf * 16 + l;
        if (nn < Ntot) {
            float bv = bias ? bias[nn] : 0.f;
#pragma unroll
            for (int v = 0; v < 8; ++v)
                C[(size_t)(mblk + half * 8 + v) * ldc + nn] = accs[nf][v] + bv;
        }
    }
}

// ---------------------------------------------------------------- GRU combine
__device__ __forceinline__ float sigm(float x) { return 1.f / (1.f + expf(-x)); }

__global__ void k_gru_combine(const float* __restrict__ gi, const float* __restrict__ gh,
                              const float* __restrict__ bih, const float* __restrict__ bhh,
                              const float* __restrict__ h, float* __restrict__ hn) {
    int i = blockIdx.x * blockDim.x + threadIdx.x;
    if (i >= cB * cH) return;
    int b = i / cH, j = i % cH;
    const float* gib = gi + b * cG;
    const float* ghb = gh + b * cG;
    float r = sigm(gib[j] + bih[j] + ghb[j] + bhh[j]);
    float z = sigm(gib[j + cH] + bih[j + cH] + ghb[j + cH] + bhh[j + cH]);
    float n = tanhf(gib[j + 2 * cH] + bih[j + 2 * cH] + r * (ghb[j + 2 * cH] + bhh[j + 2 * cH]));
    hn[i] = (1.f - z) * n + z * h[i];
}

// Layer-1 combine + masked state update + rnn_out write.
__global__ void k_gru_combine_update(
        const float* __restrict__ gi, const float* __restrict__ gh,
        const float* __restrict__ bih, const float* __restrict__ bhh,
        const float* __restrict__ h0n, float* __restrict__ h0, float* __restrict__ h1,
        const int* __restrict__ lengths, float* __restrict__ rnn_out, int t) {
    int i = blockIdx.x * blockDim.x + threadIdx.x;
    if (i >= cB * cH) return;
    int b = i / cH, j = i % cH;
    const float* gib = gi + b * cG;
    const float* ghb = gh + b * cG;
    float h1old = h1[i];
    float r = sigm(gib[j] + bih[j] + ghb[j] + bhh[j]);
    float z = sigm(gib[j + cH] + bih[j + cH] + ghb[j + cH] + bhh[j + cH]);
    float n = tanhf(gib[j + 2 * cH] + bih[j + 2 * cH] + r * (ghb[j + 2 * cH] + bhh[j + 2 * cH]));
    float h1n = (1.f - z) * n + z * h1old;
    bool m = (t < lengths[b]);
    rnn_out[((size_t)b * cT + t) * cH + j] = m ? h1n : 0.f;
    h1[i] = m ? h1n : h1old;
    if (m) h0[i] = h0n[i];
}

// ------------------------------------------------------- in-place log_softmax
__global__ void __launch_bounds__(256) k_log_softmax(float* __restrict__ out) {
    int row = blockIdx.x, tid = threadIdx.x;
    float* x = out + (size_t)row * cV;
    __shared__ float red[256];
    float mx = -3.4e38f;
    for (int i = tid; i < cV; i += 256) mx = fmaxf(mx, x[i]);
    red[tid] = mx; __syncthreads();
    for (int s = 128; s > 0; s >>= 1) {
        if (tid < s) red[tid] = fmaxf(red[tid], red[tid + s]);
        __syncthreads();
    }
    mx = red[0]; __syncthreads();
    float sum = 0.f;
    for (int i = tid; i < cV; i += 256) sum += expf(x[i] - mx);
    red[tid] = sum; __syncthreads();
    for (int s = 128; s > 0; s >>= 1) {
        if (tid < s) red[tid] += red[tid + s];
        __syncthreads();
    }
    float lse = mx + logf(red[0]);
    __syncthreads();
    for (int i = tid; i < cV; i += 256) x[i] = x[i] - lse;
}

// --------------------------------------------------------------------- driver
extern "C" void kernel_launch(void* const* d_in, const int* in_sizes, int n_in,
                              void* d_out, int out_size, void* d_ws, size_t ws_size,
                              hipStream_t stream) {
    (void)in_sizes; (void)n_in; (void)out_size; (void)ws_size;
    const float* inputs  = (const float*)d_in[0];
    const int*   lengths = (const int*)d_in[1];
    const float* memory  = (const float*)d_in[2];
    const float* state   = (const float*)d_in[3];
    const float* Wih0 = (const float*)d_in[4];
    const float* Whh0 = (const float*)d_in[5];
    const float* bih0 = (const float*)d_in[6];
    const float* bhh0 = (const float*)d_in[7];
    const float* Wih1 = (const float*)d_in[8];
    const float* Whh1 = (const float*)d_in[9];
    const float* bih1 = (const float*)d_in[10];
    const float* bhh1 = (const float*)d_in[11];
    const float* W1 = (const float*)d_in[12];
    const float* b1 = (const float*)d_in[13];
    const float* W2 = (const float*)d_in[14];
    const float* b2 = (const float*)d_in[15];
    float* out = (float*)d_out;

    // workspace carve-up (256B aligned)
    char* base = (char*)d_ws;
    size_t off = 0;
    auto take = [&](size_t bytes) -> char* {
        char* r = base + off;
        off = (off + bytes + 255) & ~(size_t)255;
        return r;
    };
    bf16* bWih0 = (bf16*)take((size_t)cG * cKX * sizeof(bf16));
    bf16* bWhh0 = (bf16*)take((size_t)cG * cH  * sizeof(bf16));
    bf16* bWih1 = (bf16*)take((size_t)cG * cH  * sizeof(bf16));
    bf16* bWhh1 = (bf16*)take((size_t)cG * cH  * sizeof(bf16));
    bf16* bW1   = (bf16*)take((size_t)cH * cH  * sizeof(bf16));
    bf16* bW2   = (bf16*)take((size_t)cVP * cH * sizeof(bf16));
    float* h0   = (float*)take((size_t)cB * cH * sizeof(float));
    float* h1   = (float*)take((size_t)cB * cH * sizeof(float));
    float* h0n  = (float*)take((size_t)cB * cH * sizeof(float));
    float* x0f  = (float*)take((size_t)cB * cKX * sizeof(float));
    float* gi0  = (float*)take((size_t)cB * cG * sizeof(float));
    float* gh0  = (float*)take((size_t)cB * cG * sizeof(float));
    float* gi1  = (float*)take((size_t)cB * cG * sizeof(float));
    float* gh1  = (float*)take((size_t)cB * cG * sizeof(float));
    float* rnn_out = (float*)take((size_t)cB * cT * cH * sizeof(float));
    float* hidden  = (float*)take((size_t)cB * cT * cH * sizeof(float));

    // one-time (per launch) weight conversion f32 -> bf16
    int n;
    n = cG * cKX; k_cvt_bf16<<<(n + 255) / 256, 256, 0, stream>>>(Wih0, bWih0, n);
    n = cG * cH;  k_cvt_bf16<<<(n + 255) / 256, 256, 0, stream>>>(Whh0, bWhh0, n);
    n = cG * cH;  k_cvt_bf16<<<(n + 255) / 256, 256, 0, stream>>>(Wih1, bWih1, n);
    n = cG * cH;  k_cvt_bf16<<<(n + 255) / 256, 256, 0, stream>>>(Whh1, bWhh1, n);
    n = cH * cH;  k_cvt_bf16<<<(n + 255) / 256, 256, 0, stream>>>(W1, bW1, n);
    n = cVP * cH; k_cvt_bf16_pad<<<(n + 255) / 256, 256, 0, stream>>>(W2, bW2, cV, cVP, cH);

    k_init_state<<<(cB * cH + 255) / 256, 256, 0, stream>>>(state, h0, h1);

    // sequential decode steps (block tile 64x64 -> grid (M/64, N/64))
    for (int t = 0; t < cT; ++t) {
        k_attention_concat<<<cB, 128, 0, stream>>>(inputs, memory, h1, x0f, t);
        k_wmma_gemm<<<dim3(1, cG / 64), 128, 0, stream>>>(x0f, cKX, bWih0, cKX, gi0, cG, cKX, cG, nullptr);
        k_wmma_gemm<<<dim3(1, cG / 64), 128, 0, stream>>>(h0,  cH,  bWhh0, cH,  gh0, cG, cH,  cG, nullptr);
        k_gru_combine<<<(cB * cH + 255) / 256, 256, 0, stream>>>(gi0, gh0, bih0, bhh0, h0, h0n);
        k_wmma_gemm<<<dim3(1, cG / 64), 128, 0, stream>>>(h0n, cH, bWih1, cH, gi1, cG, cH, cG, nullptr);
        k_wmma_gemm<<<dim3(1, cG / 64), 128, 0, stream>>>(h1,  cH, bWhh1, cH, gh1, cG, cH, cG, nullptr);
        k_gru_combine_update<<<(cB * cH + 255) / 256, 256, 0, stream>>>(
            gi1, gh1, bih1, bhh1, h0n, h0, h1, lengths, rnn_out, t);
    }

    // output head: hidden = rnn_out @ W1^T + b1 ; logits = hidden @ W2^T + b2
    k_wmma_gemm<<<dim3(cB * cT / 64, cH / 64), 128, 0, stream>>>(rnn_out, cH, bW1, cH, hidden, cH, cH, cH, b1);
    k_wmma_gemm<<<dim3(cB * cT / 64, cVP / 64), 128, 0, stream>>>(hidden, cH, bW2, cH, out, cV, cH, cV, b2);
    k_log_softmax<<<cB * cT, 256, 0, stream>>>(out);
}
